// TensorizedLoRA_52776558133759
// MI455X (gfx1250) — matrix operand order repaired and checked by reading, not compiled, not run on hardware
//
#include <hip/hip_runtime.h>

#define IN_F  4096
#define OUT_F 4096

typedef __bf16 bf16;
typedef __attribute__((ext_vector_type(16))) __bf16 v16bf;
typedef __attribute__((ext_vector_type(8)))  __bf16 v8bf;
typedef __attribute__((ext_vector_type(4)))  __bf16 v4bf;
typedef __attribute__((ext_vector_type(8)))  float  v8f;
typedef __attribute__((ext_vector_type(4)))  float  v4f;

#define WMMA_BF16(a, b, c) \
  __builtin_amdgcn_wmma_f32_16x16x32_bf16(false, (a), false, (b), (short)0, (c), false, false)

__device__ __forceinline__ void split_bf16(float x, bf16& h, bf16& l) {
  h = (bf16)x;
  l = (bf16)(x - (float)h);
}

// ---------------------------------------------------------------------------
// Kernel 1: full_weight[o,i] = weight[o,i] + prod_l s_l[i_l, o_l]
//           stored as bf16 hi/lo, (o, i) row-major (already B^T for the GEMM).
// ---------------------------------------------------------------------------
__global__ __launch_bounds__(256) void prep_weight(
    const float* __restrict__ weight,   // (OUT_F, IN_F)
    const float* __restrict__ leafs,    // (4, 8, 8, 8) [order][rank][a][b]
    bf16* __restrict__ Wh, bf16* __restrict__ Wl)
{
  __shared__ float s[256];              // s[l*64 + i_l*8 + o_l]
  const int t = threadIdx.x;
  {
    const int l = t >> 6, i = (t >> 3) & 7, j = t & 7;
    float acc = 0.f;
#pragma unroll
    for (int r = 0; r < 8; ++r)
      acc += leafs[l * 512 + r * 64 + i * 8 + j];
    s[t] = acc;
  }
  __syncthreads();

  const size_t base = (size_t)blockIdx.x * 4096;
#pragma unroll
  for (int it = 0; it < 16; ++it) {
    const size_t e = base + (size_t)it * 256 + t;
    const int o = (int)(e >> 12);
    const int i = (int)(e & 4095);
    const float tw = s[  0 + ((i >> 9) & 7) * 8 + ((o >> 9) & 7)]
                   * s[ 64 + ((i >> 6) & 7) * 8 + ((o >> 6) & 7)]
                   * s[128 + ((i >> 3) & 7) * 8 + ((o >> 3) & 7)]
                   * s[192 + ( i       & 7) * 8 + ( o       & 7)];
    const float full = weight[e] + tw;
    bf16 h, l;
    split_bf16(full, h, l);
    Wh[e] = h;
    Wl[e] = l;
  }
}

// ---------------------------------------------------------------------------
// Kernel 2: x (f32, row-major (m, k)) -> bf16 hi/lo copies.
// ---------------------------------------------------------------------------
__global__ __launch_bounds__(256) void convert_x(
    const float* __restrict__ x, bf16* __restrict__ Xh, bf16* __restrict__ Xl)
{
  const size_t idx = (size_t)blockIdx.x * 256 + threadIdx.x;   // 4 elems each
  const v4f v = ((const v4f*)x)[idx];
  v4bf h, l;
#pragma unroll
  for (int j = 0; j < 4; ++j) {
    bf16 hh, ll;
    split_bf16(v[j], hh, ll);
    h[j] = hh; l[j] = ll;
  }
  ((v4bf*)Xh)[idx] = h;
  ((v4bf*)Xl)[idx] = l;
}

// ---------------------------------------------------------------------------
// Kernel 3: C = Xf32 * W^T + bias via bf16x3 WMMA (hi*hi + hi*lo + lo*hi).
// Block: 256 thr = 8 waves (2 x 4), block tile 128x128, wave tile 64x32.
// ---------------------------------------------------------------------------
__device__ __forceinline__ v16bf load_frag(const bf16* __restrict__ p0,
                                           const bf16* __restrict__ p1) {
  const v8bf a = *(const v8bf*)p0;
  const v8bf b = *(const v8bf*)p1;
  v16bf f;
#pragma unroll
  for (int t = 0; t < 8; ++t) { f[t] = a[t]; f[t + 8] = b[t]; }
  return f;
}

__global__ __launch_bounds__(256, 1) void gemm_bf16x3(
    const bf16* __restrict__ Xh, const bf16* __restrict__ Xl,
    const bf16* __restrict__ Wh, const bf16* __restrict__ Wl,
    const float* __restrict__ bias, float* __restrict__ out)
{
  const int lane  = threadIdx.x & 31;
  const int wave  = threadIdx.x >> 5;
  const int waveM = wave >> 2;          // 0..1
  const int waveN = wave & 3;           // 0..3
  const int half  = lane >> 4;          // 0/1
  const int lrow  = lane & 15;

  const int mbase = blockIdx.y * 128 + waveM * 64;   // wave covers 64 rows
  const int nbase = blockIdx.x * 128 + waveN * 32;   // wave covers 32 cols

  // A: lane<16 -> row M=lrow, K in {0..7} u {16..23}; lane>=16 -> K+8 shift
  const bf16* Ah = Xh + (size_t)(mbase + lrow) * IN_F + half * 8;
  const bf16* Al = Xl + (size_t)(mbase + lrow) * IN_F + half * 8;
  // B: lane<16 -> col N=lrow, K in {0..7} u {16..23}; lane>=16 -> K+8 shift (W is (o,i) major)
  const bf16* Bh = Wh + (size_t)(nbase + lrow) * IN_F + half * 8;
  const bf16* Bl = Wl + (size_t)(nbase + lrow) * IN_F + half * 8;

  const v8f zero = {0.f, 0.f, 0.f, 0.f, 0.f, 0.f, 0.f, 0.f};
  v8f acc[4][2];
#pragma unroll
  for (int ms = 0; ms < 4; ++ms)
#pragma unroll
    for (int ns = 0; ns < 2; ++ns) acc[ms][ns] = zero;

#pragma unroll 1
  for (int k0 = 0; k0 < IN_F; k0 += 32) {
    v16bf ah[4], al[4], bh[2], bl[2];
#pragma unroll
    for (int ms = 0; ms < 4; ++ms) {
      const size_t ro = (size_t)ms * 16 * IN_F + k0;
      ah[ms] = load_frag(Ah + ro, Ah + ro + 16);
      al[ms] = load_frag(Al + ro, Al + ro + 16);
    }
#pragma unroll
    for (int ns = 0; ns < 2; ++ns) {
      const size_t ro = (size_t)ns * 16 * IN_F + k0;
      bh[ns] = load_frag(Bh + ro, Bh + ro + 16);
      bl[ns] = load_frag(Bl + ro, Bl + ro + 16);
    }
    // WGP-scope prefetch of the next K chunks (locality 3 -> all cache levels)
    __builtin_prefetch(Ah + k0 + 64, 0, 3);
    __builtin_prefetch(Al + k0 + 64, 0, 3);
    __builtin_prefetch(Bh + k0 + 64, 0, 3);
    __builtin_prefetch(Bl + k0 + 64, 0, 3);

#pragma unroll
    for (int ms = 0; ms < 4; ++ms)
#pragma unroll
      for (int ns = 0; ns < 2; ++ns) {
        acc[ms][ns] = WMMA_BF16(ah[ms], bh[ns], acc[ms][ns]);  // hi*hi
        acc[ms][ns] = WMMA_BF16(ah[ms], bl[ns], acc[ms][ns]);  // hi*lo
        acc[ms][ns] = WMMA_BF16(al[ms], bh[ns], acc[ms][ns]);  // lo*hi
      }
  }

  // C/D layout: VGPR r, lane<16 -> (M=r, N=lane); lane>=16 -> (M=8+r, N=lane-16)
#pragma unroll
  for (int ms = 0; ms < 4; ++ms)
#pragma unroll
    for (int ns = 0; ns < 2; ++ns) {
      const int orow = mbase + ms * 16 + half * 8;
      const int ocol = nbase + ns * 16 + lrow;
      const float bv = bias[ocol];
#pragma unroll
      for (int r = 0; r < 8; ++r)
        out[(size_t)(orow + r) * OUT_F + ocol] = acc[ms][ns][r] + bv;
    }
}

// ---------------------------------------------------------------------------
extern "C" void kernel_launch(void* const* d_in, const int* in_sizes, int n_in,
                              void* d_out, int out_size, void* d_ws, size_t ws_size,
                              hipStream_t stream) {
  const float* x      = (const float*)d_in[0];   // (8, 2048, 4096)
  const float* weight = (const float*)d_in[1];   // (4096, 4096)
  const float* bias   = (const float*)d_in[2];   // (4096,)
  const float* leafs  = (const float*)d_in[3];   // (4, 8, 8, 8)
  float* out = (float*)d_out;

  const int    M  = in_sizes[0] / IN_F;          // 16384
  const size_t WE = (size_t)OUT_F * IN_F;        // 16.7M
  const size_t XE = (size_t)M * IN_F;            // 67.1M

  bf16* Wh = (bf16*)d_ws;
  bf16* Wl = Wh + WE;
  bf16* Xh = Wl + WE;
  bf16* Xl = Xh + XE;

  prep_weight<<<dim3((unsigned)(WE / 4096)), 256, 0, stream>>>(weight, leafs, Wh, Wl);
  convert_x  <<<dim3((unsigned)(XE / 1024)), 256, 0, stream>>>(x, Xh, Xl);
  gemm_bf16x3<<<dim3(OUT_F / 128, M / 128), 256, 0, stream>>>(Xh, Xl, Wh, Wl, bias, out);
}
